// CrossMultiheadAttention_44074954391814
// MI455X (gfx1250) — matrix operand order repaired and checked by reading, not compiled, not run on hardware
//
// CDNA5 / gfx1250 cross-multihead-attention.
// All six GEMMs on v_wmma_f32_16x16x32_bf16 (wave32, f32 accumulate).
// Register-blocked (16x64 per wave) + software-pipelined operand loads so
// 4 WMMAs issue back-to-back per K-step. q-tiles staged into LDS with
// global_load_async_to_lds_b128 (ASYNCcnt path). Working set (<100MB) is
// L2-resident on MI455X (192MB L2, 23.3TB/s HBM).
#include <hip/hip_runtime.h>

#define D_MODEL 1024
#define NUM_HEAD 16
#define HEAD_DIM 64
#define BATCH 4
#define SEQ 1024

typedef __bf16 bf16;
typedef __attribute__((ext_vector_type(16))) __bf16 v16bf;
typedef __attribute__((ext_vector_type(8)))  __bf16 v8bf;
typedef __attribute__((ext_vector_type(8)))  float  v8f;

// ---------------------------------------------------------------- conversion
__global__ void cvt_f32_to_bf16(const float* __restrict__ in,
                                bf16* __restrict__ out, int n) {
  int i = blockIdx.x * blockDim.x + threadIdx.x;
  int stride = gridDim.x * blockDim.x;
  for (; i < n; i += stride) out[i] = (bf16)in[i];
}

// A-operand (16x32 bf16): lane holds row (l%16); lanes 0-15 take K {0..15},
// lanes 16-31 take K {16..31}  (contiguous, matches B loads).
__device__ __forceinline__ v16bf load_a_bf16(const bf16* rowbase, int half) {
  const v8bf lo = *(const v8bf*)(rowbase + half * 16);
  const v8bf hi = *(const v8bf*)(rowbase + half * 16 + 8);
  v16bf a;
#pragma unroll
  for (int i = 0; i < 8; ++i) { a[i] = lo[i]; a[i + 8] = hi[i]; }
  return a;
}

__device__ __forceinline__ v8f wmma_bf16(v16bf a, v16bf b, v8f c) {
  return __builtin_amdgcn_wmma_f32_16x16x32_bf16(false, a, false, b,
                                                 (short)0, c, false, false);
}

// Per-lane async 16B copy global -> LDS (gfx1250 ASYNCcnt path).
__device__ __forceinline__ void async_copy_b128(unsigned lds_byte_off,
                                                const bf16* gaddr) {
  asm volatile("global_load_async_to_lds_b128 %0, %1, off"
               :: "v"(lds_byte_off), "v"(gaddr) : "memory");
}
__device__ __forceinline__ void wait_async0() {
  asm volatile("s_wait_asynccnt 0x0" ::: "memory");
}

// -------------------------------------------------------------------- GEMM
// C[m,n] = sum_k A[m,k] * W[n,k] + bias[n].  One wave owns a 16x64 C block:
// one A fetch feeds 4 WMMAs; next K-step operands prefetched into registers.
// MODE 0: bf16 row-major out; MODE 1: bf16 out transposed to (B,H,hd,S);
// MODE 2: f32 row-major out.
template <int MODE>
__global__ void __launch_bounds__(128)
gemm_bias_16x64(const bf16* __restrict__ A, const bf16* __restrict__ W,
                const float* __restrict__ bias, void* __restrict__ outv,
                int M, int N, int K) {
  const int lane = threadIdx.x & 31;
  const int wave = threadIdx.x >> 5;       // 0..3
  const int half = lane >> 4;
  const int l16  = lane & 15;
  const int m0 = blockIdx.x * 16;
  const int n0 = blockIdx.y * 256 + wave * 64;
  const int boff = half * 16;

  const bf16* arow  = A + (size_t)(m0 + l16) * K;
  const bf16* brow0 = W + (size_t)(n0 +  0 + l16) * K;
  const bf16* brow1 = W + (size_t)(n0 + 16 + l16) * K;
  const bf16* brow2 = W + (size_t)(n0 + 32 + l16) * K;
  const bf16* brow3 = W + (size_t)(n0 + 48 + l16) * K;

  v8f acc0 = {}, acc1 = {}, acc2 = {}, acc3 = {};
  v16bf a  = load_a_bf16(arow, half);
  v16bf b0 = *(const v16bf*)(brow0 + boff);
  v16bf b1 = *(const v16bf*)(brow1 + boff);
  v16bf b2 = *(const v16bf*)(brow2 + boff);
  v16bf b3 = *(const v16bf*)(brow3 + boff);

  for (int k0 = 0; k0 < K; k0 += 32) {
    const int kn = (k0 + 32 < K) ? (k0 + 32) : 0;   // clamp: loads stay unconditional
    __builtin_prefetch(brow0 + kn + 512, 0, 0);     // global_prefetch_b8
    v16bf an  = load_a_bf16(arow + kn, half);
    v16bf bn0 = *(const v16bf*)(brow0 + kn + boff);
    v16bf bn1 = *(const v16bf*)(brow1 + kn + boff);
    v16bf bn2 = *(const v16bf*)(brow2 + kn + boff);
    v16bf bn3 = *(const v16bf*)(brow3 + kn + boff);
    acc0 = wmma_bf16(a, b0, acc0);
    acc1 = wmma_bf16(a, b1, acc1);
    acc2 = wmma_bf16(a, b2, acc2);
    acc3 = wmma_bf16(a, b3, acc3);
    a = an; b0 = bn0; b1 = bn1; b2 = bn2; b3 = bn3;
  }

  v8f accs[4] = {acc0, acc1, acc2, acc3};
#pragma unroll
  for (int j = 0; j < 4; ++j) {
    const int n = n0 + j * 16 + l16;
    const float bn = bias[n];
    if (MODE == 0) {
      bf16* out = (bf16*)outv;
#pragma unroll
      for (int g = 0; g < 8; ++g) {
        int m = m0 + half * 8 + g;
        out[(size_t)m * N + n] = (bf16)(accs[j][g] + bn);
      }
    } else if (MODE == 1) {
      // V transposed: vt[b][h][d][s]; 8 consecutive s per lane -> one 16B store
      bf16* out = (bf16*)outv;
      const int b  = m0 / SEQ;
      const int s0 = (m0 % SEQ) + half * 8;
      const int hh = n / HEAD_DIM;
      const int d  = n % HEAD_DIM;
      v8bf pk;
#pragma unroll
      for (int g = 0; g < 8; ++g) pk[g] = (bf16)(accs[j][g] + bn);
      *(v8bf*)(out + ((size_t)((b * NUM_HEAD + hh) * HEAD_DIM + d)) * SEQ + s0) = pk;
    } else {
      float* out = (float*)outv;
#pragma unroll
      for (int g = 0; g < 8; ++g) {
        int m = m0 + half * 8 + g;
        out[(size_t)m * N + n] = accs[j][g] + bn;
      }
    }
  }
}

// ------------------------------------------------------- fused attention core
// One block per (head h, 16-row q-tile). 128 threads = 4 waves.
// LDS: scores 16x1024 f32 | acc 16x1024 f32 | reduction scratch | q-tile bf16.
// Per batch b: q-tile async-staged to LDS, WMMA scores -> LDS, row softmax,
// accumulate over b (ref sums softmax over batch). Then acc converted once to
// bf16 in LDS and attn @ v^T runs with ds_load A operands.
#define LDS_SCORES   0
#define LDS_ACC      65536
#define LDS_RED      131072
#define LDS_QTILE    131584
#define LDS_TOTAL    133632

__global__ void __launch_bounds__(128)
attn_fused(const bf16* __restrict__ qb, const bf16* __restrict__ kb,
           const bf16* __restrict__ vtb, const float* __restrict__ mask,
           bf16* __restrict__ Vout) {
  extern __shared__ char smemraw[];
  float* scores = (float*)(smemraw + LDS_SCORES);   // 16*1024 f32
  float* accp   = (float*)(smemraw + LDS_ACC);      // 16*1024 f32
  float* red    = (float*)(smemraw + LDS_RED);      // 16*8 f32
  bf16*  qtile  = (bf16*)(smemraw + LDS_QTILE);     // 16*64 bf16
  bf16*  attnb  = (bf16*)(smemraw + LDS_SCORES);    // reuses scores after b-loop

  const int h   = blockIdx.x;   // 0..15
  const int qt  = blockIdx.y;   // 0..63
  const int q0  = qt * 16;
  const int tid = threadIdx.x;
  const int lane = tid & 31;
  const int wave = tid >> 5;
  const int half = lane >> 4;
  const int l16  = lane & 15;

  for (int i = tid; i < 16 * 1024; i += 128) accp[i] = 0.0f;
  __syncthreads();

  const float scale = 0.125f;  // 1/sqrt(HEAD_DIM)

  for (int b = 0; b < BATCH; ++b) {
    // ---- stage q-tile (16x64 bf16 = 2KB) into LDS via async copy
    {
      const int row = tid >> 3;      // 0..15
      const int ch  = tid & 7;       // 8 bf16 (16B) chunks per row
      const bf16* g = qb + ((size_t)(b * SEQ + q0 + row) * D_MODEL
                            + h * HEAD_DIM + ch * 8);
      async_copy_b128((unsigned)(LDS_QTILE + tid * 16), g);
      wait_async0();
    }
    __syncthreads();

    // A operands (invariant over column tiles): d0 = 0 and d0 = 32
    const bf16* qrow_l = qtile + l16 * HEAD_DIM;
    const v16bf a_lo = load_a_bf16(qrow_l, half);
    const v16bf a_hi = load_a_bf16(qrow_l + 32, half);

    // ---- scores: wave owns k-columns [wave*256, wave*256+256), pipelined B
    const bf16* kbase =
        kb + ((size_t)(b * SEQ + wave * 256 + l16) * D_MODEL + h * HEAD_DIM);
    const size_t krstride = (size_t)16 * D_MODEL;   // 16 k-rows per column tile
    v16bf blo = *(const v16bf*)(kbase + half * 16);
    v16bf bhi = *(const v16bf*)(kbase + 32 + half * 16);
    for (int ct = 0; ct < 16; ++ct) {
      const int ctn = (ct + 1) & 15;
      const bf16* knext = kbase + (size_t)ctn * krstride;
      v16bf nlo = *(const v16bf*)(knext + half * 16);
      v16bf nhi = *(const v16bf*)(knext + 32 + half * 16);
      v8f acc = {};
      acc = wmma_bf16(a_lo, blo, acc);
      acc = wmma_bf16(a_hi, bhi, acc);
      const int c0 = wave * 256 + ct * 16;
#pragma unroll
      for (int g = 0; g < 8; ++g) {
        const int row = half * 8 + g;
        const int col = c0 + l16;
        scores[row * 1024 + col] =
            acc[g] * scale + mask[(size_t)(q0 + row) * SEQ + col];
      }
      blo = nlo; bhi = nhi;
    }
    __syncthreads();

    // ---- softmax per row: 8 threads per row
    const int row = tid >> 3;
    const int sub = tid & 7;
    float m = -3.402823466e38f;
    for (int c = sub * 128; c < sub * 128 + 128; ++c)
      m = fmaxf(m, scores[row * 1024 + c]);
    red[row * 8 + sub] = m;
    __syncthreads();
    float rowmax = red[row * 8];
#pragma unroll
    for (int i = 1; i < 8; ++i) rowmax = fmaxf(rowmax, red[row * 8 + i]);
    float s = 0.0f;
    for (int c = sub * 128; c < sub * 128 + 128; ++c) {
      float e = __expf(scores[row * 1024 + c] - rowmax);
      scores[row * 1024 + c] = e;
      s += e;
    }
    __syncthreads();
    red[row * 8 + sub] = s;
    __syncthreads();
    float rowsum = 0.0f;
#pragma unroll
    for (int i = 0; i < 8; ++i) rowsum += red[row * 8 + i];
    const float inv = 1.0f / rowsum;
    for (int c = sub * 128; c < sub * 128 + 128; ++c)
      accp[row * 1024 + c] += scores[row * 1024 + c] * inv;
    __syncthreads();
  }

  // ---- convert accumulator once to bf16 (reuses scores LDS region)
  for (int i = tid; i < 16 * 1024; i += 128) attnb[i] = (bf16)accp[i];
  __syncthreads();

  // ---- attn(H,Sq,Sk tile) @ v^T : wave owns d-tile [wave*16, +16), pipelined B
  const int dcol = wave * 16 + l16;
  const bf16* arow_l = attnb + l16 * 1024;   // A from LDS (ds_load_b128 pairs)
  for (int b = 0; b < BATCH; ++b) {
    const bf16* vrow =
        vtb + ((size_t)((b * NUM_HEAD + h) * HEAD_DIM + dcol)) * SEQ;
    v8f acc = {};
    v16bf bv = *(const v16bf*)(vrow + half * 16);
    for (int k0 = 0; k0 < SEQ; k0 += 32) {
      const int kn = (k0 + 32 < SEQ) ? (k0 + 32) : 0;
      v16bf bnv = *(const v16bf*)(vrow + kn + half * 16);
      v16bf a = load_a_bf16(arow_l + k0, half);
      acc = wmma_bf16(a, bv, acc);
      bv = bnv;
    }
#pragma unroll
    for (int g = 0; g < 8; ++g) {
      const int s = q0 + half * 8 + g;
      Vout[(size_t)(b * SEQ + s) * D_MODEL + h * HEAD_DIM + dcol] = (bf16)acc[g];
    }
  }
}

// ------------------------------------------------------------------- launch
extern "C" void kernel_launch(void* const* d_in, const int* in_sizes, int n_in,
                              void* d_out, int out_size, void* d_ws, size_t ws_size,
                              hipStream_t stream) {
  const float* x    = (const float*)d_in[0];
  const float* y    = (const float*)d_in[1];
  const float* mask = (const float*)d_in[2];
  const float* Wq   = (const float*)d_in[3];
  const float* bq   = (const float*)d_in[4];
  const float* Wk   = (const float*)d_in[5];
  const float* bk   = (const float*)d_in[6];
  const float* Wv   = (const float*)d_in[7];
  const float* bv   = (const float*)d_in[8];
  const float* Wo   = (const float*)d_in[9];
  const float* bo   = (const float*)d_in[10];
  float* out = (float*)d_out;

  const size_t NX = (size_t)BATCH * SEQ * D_MODEL;  // 4,194,304
  const size_t NW = (size_t)D_MODEL * D_MODEL;      // 1,048,576

  bf16* ws  = (bf16*)d_ws;
  bf16* xb  = ws;  ws += NX;
  bf16* yb  = ws;  ws += NX;
  bf16* Wqb = ws;  ws += NW;
  bf16* Wkb = ws;  ws += NW;
  bf16* Wvb = ws;  ws += NW;
  bf16* Wob = ws;  ws += NW;
  bf16* qb  = ws;  ws += NX;
  bf16* kb  = ws;  ws += NX;
  bf16* vtb = ws;  ws += NX;   // transposed: (B,H,hd,S)
  bf16* Vb  = ws;  ws += NX;   // (B,S,H*hd)

  cvt_f32_to_bf16<<<1024, 256, 0, stream>>>(x,  xb,  (int)NX);
  cvt_f32_to_bf16<<<1024, 256, 0, stream>>>(y,  yb,  (int)NX);
  cvt_f32_to_bf16<<<512,  256, 0, stream>>>(Wq, Wqb, (int)NW);
  cvt_f32_to_bf16<<<512,  256, 0, stream>>>(Wk, Wkb, (int)NW);
  cvt_f32_to_bf16<<<512,  256, 0, stream>>>(Wv, Wvb, (int)NW);
  cvt_f32_to_bf16<<<512,  256, 0, stream>>>(Wo, Wob, (int)NW);

  const int M = BATCH * SEQ;  // 4096
  dim3 gg(M / 16, D_MODEL / 256);  // (256, 4), 128 thr = 4 waves/block
  gemm_bias_16x64<0><<<gg, 128, 0, stream>>>(yb, Wqb, bq, qb,  M, D_MODEL, D_MODEL);
  gemm_bias_16x64<0><<<gg, 128, 0, stream>>>(xb, Wkb, bk, kb,  M, D_MODEL, D_MODEL);
  gemm_bias_16x64<1><<<gg, 128, 0, stream>>>(xb, Wvb, bv, vtb, M, D_MODEL, D_MODEL);

  // ~130.5 KB dynamic LDS: legal on gfx1250 (320 KB per workgroup).
  dim3 ga(NUM_HEAD, SEQ / 16);  // (16, 64)
  attn_fused<<<ga, 128, (size_t)LDS_TOTAL, stream>>>(qb, kb, vtb, mask, Vb);

  gemm_bias_16x64<2><<<gg, 128, 0, stream>>>(Vb, Wob, bo, out, M, D_MODEL, D_MODEL);
}
